// ParT_LHAASO_Small_4561255268374
// MI455X (gfx1250) — matrix operand order repaired and checked by reading, not compiled, not run on hardware
//
#include <hip/hip_runtime.h>
#include <hip/hip_bf16.h>
#include <math.h>

// ---------------------------------------------------------------------------
// ParT-small forward for MI455X (gfx1250, wave32, WMMA).
// All heavy contractions use v_wmma_f32_16x16x32_f16 (f16 in, f32 acc).
// Shapes: B=4 N=512 IN=5 D=128 H=8 HD=16 L=4
// ---------------------------------------------------------------------------

#define BN   4
#define NN   512
#define IND  5
#define DD   128
#define HH   8
#define HD   16
#define LL   4
#define FF   512   // 4*D

typedef __attribute__((ext_vector_type(16))) _Float16 v16h;
typedef __attribute__((ext_vector_type(8)))  _Float16 h8;
typedef __attribute__((ext_vector_type(8)))  float    v8f;

static __device__ __forceinline__ float gelu_exact(float x) {
    return 0.5f * x * (1.0f + erff(x * 0.70710678118654752f));
}

// ---- WMMA fragment helpers (ISA 7.12.2 layouts, wave32) -------------------
// A (16x32 f16, row-major source, stride lda halves, 16B-aligned rows):
//   lanes 0-15: regs0-3 = K0..7, regs4-7 = K16..23 ; lanes16-31: K8..15 / K24..31
static __device__ __forceinline__ v16h load_a_frag(const _Float16* __restrict__ A,
                                                   int lda, int lane) {
    const _Float16* p = A + (lane & 15) * lda + ((lane & 16) ? 8 : 0);
    h8 lo = *(const h8*)p;
    h8 hi = *(const h8*)(p + 16);
    v16h a;
#pragma unroll
    for (int j = 0; j < 8; ++j) { a[j] = lo[j]; a[j + 8] = hi[j]; }
    return a;
}
// B (32x16 f16) loaded from an N-major transposed matrix Bt[n][k]:
//   lanes 0-15: K0..7, K16..23 of column n=lane ; lanes16-31: K8..15 / K24..31.
static __device__ __forceinline__ v16h load_b_frag(const _Float16* __restrict__ Bt,
                                                   int ldb, int lane) {
    const _Float16* p = Bt + (lane & 15) * ldb + ((lane & 16) ? 8 : 0);
    h8 lo = *(const h8*)p;
    h8 hi = *(const h8*)(p + 16);
    v16h b;
#pragma unroll
    for (int j = 0; j < 8; ++j) { b[j] = lo[j]; b[j + 8] = hi[j]; }
    return b;
}

// ---- block reduction helpers ----------------------------------------------
static __device__ __forceinline__ float block_sum(float v, float* sm, int nthr) {
    int t = threadIdx.x;
    sm[t] = v; __syncthreads();
    for (int s = nthr >> 1; s > 0; s >>= 1) {
        if (t < s) sm[t] += sm[t + s];
        __syncthreads();
    }
    float r = sm[0]; __syncthreads();
    return r;
}
static __device__ __forceinline__ float block_max(float v, float* sm, int nthr) {
    int t = threadIdx.x;
    sm[t] = v; __syncthreads();
    for (int s = nthr >> 1; s > 0; s >>= 1) {
        if (t < s) sm[t] = fmaxf(sm[t], sm[t + s]);
        __syncthreads();
    }
    float r = sm[0]; __syncthreads();
    return r;
}

// ===========================================================================
// Prep kernels
// ===========================================================================

// f32 W(K,N) -> f16 Wt(N,K)
__global__ void conv_transpose_kernel(const float* __restrict__ W,
                                      _Float16* __restrict__ Wt, int K, int N) {
    int idx = blockIdx.x * 256 + threadIdx.x;
    if (idx >= K * N) return;
    int n = idx / K, k = idx - n * K;
    Wt[idx] = (_Float16)W[k * N + n];
}

// pm_W2 (L,128,8) f32 -> (L,16,128) f16, head rows padded 8->16 with zeros
__global__ void pmw2t_kernel(const float* __restrict__ W2, _Float16* __restrict__ Wt) {
    int idx = blockIdx.x * 256 + threadIdx.x;   // L*16*128 = 8192
    if (idx >= LL * 16 * DD) return;
    int col = idx & (DD - 1);
    int row = (idx >> 7) & 15;
    int l   = idx >> 11;
    Wt[idx] = (row < HH) ? (_Float16)W2[l * DD * HH + col * HH + row] : (_Float16)0.f;
}

// embedding + LN + gelu + positional gelu -> feat f32 & f16
__global__ __launch_bounds__(128)
void embed_kernel(const float* __restrict__ hits,
                  const float* __restrict__ eW, const float* __restrict__ eb,
                  const float* __restrict__ eg, const float* __restrict__ ebe,
                  const float* __restrict__ pW, const float* __restrict__ pb,
                  float* __restrict__ feat, _Float16* __restrict__ feat16) {
    __shared__ float sm[128];
    int row = blockIdx.x;            // b*N + n
    int t = threadIdx.x;
    const float* hp = hits + row * IND;
    float e = eb[t];
#pragma unroll
    for (int j = 0; j < IND; ++j) e += hp[j] * eW[j * DD + t];
    float mean = block_sum(e, sm, 128) * (1.0f / DD);
    float d = e - mean;
    float var = block_sum(d * d, sm, 128) * (1.0f / DD);
    float xn = d * rsqrtf(var + 1e-5f) * eg[t] + ebe[t];
    float pv = hp[0] * pW[t] + hp[1] * pW[DD + t] + pb[t];
    float f = gelu_exact(xn) + gelu_exact(pv);
    feat[row * DD + t] = f;
    feat16[row * DD + t] = (_Float16)f;
}

// mean(|dd|) denominators per (b, coord)
__global__ __launch_bounds__(256)
void pair_den_kernel(const float* __restrict__ hits, float* __restrict__ den) {
    __shared__ float sm[256];
    int b = blockIdx.x / 3, c = blockIdx.x % 3;
    int t = threadIdx.x;
    float s = 0.f;
    for (int ij = t; ij < NN * NN; ij += 256) {
        int i = ij >> 9, j = ij & (NN - 1);
        s += fabsf(hits[(b * NN + i) * IND + c] - hits[(b * NN + j) * IND + c]);
    }
    float tot = block_sum(s, sm, 256);
    if (t == 0) den[blockIdx.x] = tot * (1.0f / (NN * NN)) + 1e-6f;
}

// pair2(b,i,j,0..1) = pd3 @ pp_W + pp_b
__global__ __launch_bounds__(256)
void pair2_kernel(const float* __restrict__ hits, const float* __restrict__ den,
                  const float* __restrict__ ppW, const float* __restrict__ ppb,
                  float* __restrict__ pair2) {
    int idx = blockIdx.x * 256 + threadIdx.x;   // B*N*N
    int j = idx & (NN - 1);
    int i = (idx >> 9) & (NN - 1);
    int b = idx >> 18;
    float o0 = ppb[0], o1 = ppb[1];
#pragma unroll
    for (int c = 0; c < 3; ++c) {
        float dd = (hits[(b * NN + i) * IND + c] - hits[(b * NN + j) * IND + c])
                   / den[b * 3 + c];
        o0 += dd * ppW[c * 2 + 0];
        o1 += dd * ppW[c * 2 + 1];
    }
    pair2[(size_t)idx * 2 + 0] = o0;
    pair2[(size_t)idx * 2 + 1] = o1;
}

// ===========================================================================
// Generic WMMA GEMM:  C(M,N) = A(M,K) @ B(K,N), A f16 row-major, Bt f16 (N,K)
// ===========================================================================
enum { EPI_F16 = 0, EPI_BIAS_F32 = 1, EPI_BIAS_GELU_F16 = 2 };

template <int EPI>
__global__ __launch_bounds__(32)
void wmma_gemm_kernel(const _Float16* __restrict__ A, const _Float16* __restrict__ Bt,
                      const float* __restrict__ bias, void* __restrict__ Cout,
                      int K, int lda, int ldb, int ldc) {
    int lane = threadIdx.x;
    const _Float16* Ab = A + (size_t)(blockIdx.y * 16) * lda;
    const _Float16* Bb = Bt + (size_t)(blockIdx.x * 16) * ldb;
    v8f acc = {};
    for (int k0 = 0; k0 < K; k0 += 32) {
        v16h a = load_a_frag(Ab + k0, lda, lane);
        v16h b = load_b_frag(Bb + k0, ldb, lane);
        acc = __builtin_amdgcn_wmma_f32_16x16x32_f16(false, a, false, b,
                                                     (short)0, acc, false, false);
    }
    int n = blockIdx.x * 16 + (lane & 15);
    int mbase = blockIdx.y * 16 + ((lane & 16) ? 8 : 0);
#pragma unroll
    for (int r = 0; r < 8; ++r) {
        int m = mbase + r;
        float v = acc[r];
        if (EPI == EPI_F16) {
            ((_Float16*)Cout)[(size_t)m * ldc + n] = (_Float16)v;
        } else if (EPI == EPI_BIAS_F32) {
            ((float*)Cout)[(size_t)m * ldc + n] = v + bias[n];
        } else {
            ((_Float16*)Cout)[(size_t)m * ldc + n] = (_Float16)gelu_exact(v + bias[n]);
        }
    }
}

// ===========================================================================
// Per-layer kernels
// ===========================================================================

// fused pair-bias MLP: hidden built in VALU straight into the A fragment,
// then bias(b,q,k,h) = relu(pair2 @ W1 + b1) @ W2 + b2 via 4 chained WMMAs.
__global__ __launch_bounds__(32)
void bias_kernel(const float* __restrict__ pair2,
                 const float* __restrict__ W1, const float* __restrict__ b1,
                 const _Float16* __restrict__ W2t, const float* __restrict__ b2,
                 _Float16* __restrict__ bias16) {
    int kt = blockIdx.x, q = blockIdx.y, b = blockIdx.z;
    int lane = threadIdx.x;
    int m = lane & 15;
    int ko = (lane & 16) ? 8 : 0;
    const float* p2 = pair2 + (((size_t)b * NN + q) * NN + kt * 16 + m) * 2;
    float p0 = p2[0], p1 = p2[1];
    v8f acc = {};
#pragma unroll
    for (int c = 0; c < 4; ++c) {
        v16h a;
#pragma unroll
        for (int j = 0; j < 8; ++j) {
            int ch = c * 32 + ko + j;
            a[j] = (_Float16)fmaxf(0.f, p0 * W1[ch] + p1 * W1[DD + ch] + b1[ch]);
        }
#pragma unroll
        for (int j = 0; j < 8; ++j) {
            int ch = c * 32 + ko + 16 + j;
            a[j + 8] = (_Float16)fmaxf(0.f, p0 * W1[ch] + p1 * W1[DD + ch] + b1[ch]);
        }
        v16h bf = load_b_frag(W2t + c * 32, DD, lane);
        acc = __builtin_amdgcn_wmma_f32_16x16x32_f16(false, a, false, bf,
                                                     (short)0, acc, false, false);
    }
    int n = lane & 15;               // head column (0..15, only 0..7 valid)
    int kbase = kt * 16 + ((lane & 16) ? 8 : 0);
    if (n < HH) {
#pragma unroll
        for (int r = 0; r < 8; ++r) {
            int kk = kbase + r;
            bias16[(((size_t)b * NN + q) * NN + kk) * HH + n] =
                (_Float16)(acc[r] + b2[n]);
        }
    }
}

// logits(b,h,q,k) = (Q·K^T)*scale + bias ; mask -> -1e9   (K=HD=16 padded to 32)
__global__ __launch_bounds__(32)
void logits_kernel(const _Float16* __restrict__ qkv,
                   const _Float16* __restrict__ bias16,
                   const unsigned char* __restrict__ mask,
                   float* __restrict__ logits) {
    const float scale = 0.25f;   // HD^-0.5
    int kt = blockIdx.x, qt = blockIdx.y;
    int b = blockIdx.z >> 3, h = blockIdx.z & 7;
    int lane = threadIdx.x;
    // A fragment = Q rows (channels 0..15 real, 16..31 zero)
    int m = lane & 15;
    int ko = (lane & 16) ? 8 : 0;
    const _Float16* qp = qkv + ((size_t)(b * NN + qt * 16 + m)) * (3 * DD) + h * HD + ko;
    v16h a = {};
    {
        h8 lo = *(const h8*)qp;
#pragma unroll
        for (int j = 0; j < 8; ++j) a[j] = lo[j];
    }
    // B fragment = K rows (channels 0..15 real, 16..31 zero)
    int n = lane & 15;
    v16h bf = {};
    {
        const _Float16* kp = qkv + ((size_t)(b * NN + kt * 16 + n)) * (3 * DD) + DD + h * HD + ko;
        h8 lo = *(const h8*)kp;
#pragma unroll
        for (int j = 0; j < 8; ++j) bf[j] = lo[j];
    }
    v8f acc = {};
    acc = __builtin_amdgcn_wmma_f32_16x16x32_f16(false, a, false, bf,
                                                 (short)0, acc, false, false);
    int kk = kt * 16 + n;
    int qbase = qt * 16 + ((lane & 16) ? 8 : 0);
    bool ok = mask[b * NN + kk] != 0;
#pragma unroll
    for (int r = 0; r < 8; ++r) {
        int q = qbase + r;
        float v = acc[r] * scale +
                  (float)bias16[(((size_t)b * NN + q) * NN + kk) * HH + h];
        if (!ok) v = -1e9f;
        logits[(((size_t)(b * HH + h) * NN) + q) * NN + kk] = v;
    }
}

// softmax over the key dim, writes f16 attention
__global__ __launch_bounds__(128)
void softmax_kernel(const float* __restrict__ logits, _Float16* __restrict__ attn16) {
    __shared__ float sm[128];
    size_t base = (size_t)blockIdx.x * NN;
    int t = threadIdx.x;
    float v[4];
    float mx = -3.402823e38f;
#pragma unroll
    for (int j = 0; j < 4; ++j) { v[j] = logits[base + t + j * 128]; mx = fmaxf(mx, v[j]); }
    mx = block_max(mx, sm, 128);
    float s = 0.f;
#pragma unroll
    for (int j = 0; j < 4; ++j) { v[j] = expf(v[j] - mx); s += v[j]; }
    s = block_sum(s, sm, 128);
    float inv = 1.0f / s;
#pragma unroll
    for (int j = 0; j < 4; ++j) attn16[base + t + j * 128] = (_Float16)(v[j] * inv);
}

// V transpose: vt(b,h,ch,key) f16 from qkv
__global__ __launch_bounds__(256)
void vt_kernel(const _Float16* __restrict__ qkv, _Float16* __restrict__ vt) {
    int idx = blockIdx.x * 256 + threadIdx.x;     // B*H*HD*N
    int key = idx & (NN - 1);
    int ch  = (idx >> 9) & 15;
    int h   = (idx >> 13) & 7;
    int b   = idx >> 16;
    vt[idx] = qkv[((size_t)(b * NN + key)) * (3 * DD) + 2 * DD + h * HD + ch];
}

// o(b,q,h*16+ch) = attn @ V   (K=512, 16 WMMAs per tile)
__global__ __launch_bounds__(32)
void av_kernel(const _Float16* __restrict__ attn16, const _Float16* __restrict__ vt,
               _Float16* __restrict__ o16) {
    int qt = blockIdx.x;
    int b = blockIdx.y >> 3, h = blockIdx.y & 7;
    int lane = threadIdx.x;
    const _Float16* Ab = attn16 + ((size_t)(b * HH + h) * NN + qt * 16) * NN;
    const _Float16* Bb = vt + ((size_t)(b * HH + h) * HD) * NN;
    v8f acc = {};
    for (int k0 = 0; k0 < NN; k0 += 32) {
        v16h a = load_a_frag(Ab + k0, NN, lane);
        v16h bf = load_b_frag(Bb + k0, NN, lane);
        acc = __builtin_amdgcn_wmma_f32_16x16x32_f16(false, a, false, bf,
                                                     (short)0, acc, false, false);
    }
    int n = lane & 15;
    int qbase = qt * 16 + ((lane & 16) ? 8 : 0);
#pragma unroll
    for (int r = 0; r < 8; ++r) {
        o16[((size_t)(b * NN) + qbase + r) * DD + h * HD + n] = (_Float16)acc[r];
    }
}

// feat = LN(feat + tmp) ; also refresh f16 copy
__global__ __launch_bounds__(128)
void ln_kernel(float* __restrict__ feat, _Float16* __restrict__ feat16,
               const float* __restrict__ tmp,
               const float* __restrict__ g, const float* __restrict__ bb) {
    __shared__ float sm[128];
    int row = blockIdx.x;
    int t = threadIdx.x;
    float x = feat[(size_t)row * DD + t] + tmp[(size_t)row * DD + t];
    float mean = block_sum(x, sm, 128) * (1.0f / DD);
    float d = x - mean;
    float var = block_sum(d * d, sm, 128) * (1.0f / DD);
    float v = d * rsqrtf(var + 1e-5f) * g[t] + bb[t];
    feat[(size_t)row * DD + t] = v;
    feat16[(size_t)row * DD + t] = (_Float16)v;
}

// masked mean pool + param branch + classifier head (tiny, VALU)
__global__ __launch_bounds__(128)
void head_kernel(const float* __restrict__ feat, const unsigned char* __restrict__ mask,
                 const float* __restrict__ params,
                 const float* __restrict__ pfW, const float* __restrict__ pfb,
                 const float* __restrict__ pfg, const float* __restrict__ pfbe,
                 const float* __restrict__ c1W, const float* __restrict__ c1b,
                 const float* __restrict__ c2W, const float* __restrict__ c2b,
                 float* __restrict__ out) {
    __shared__ float sm[128];
    __shared__ float cat[2 * DD];
    __shared__ float h1[DD];
    int b = blockIdx.x, t = threadIdx.x;
    float s = 0.f, cnt = 0.f;
    for (int n = 0; n < NN; ++n) {
        float mf = mask[b * NN + n] ? 1.0f : 0.0f;
        s += feat[((size_t)b * NN + n) * DD + t] * mf;
        cnt += mf;
    }
    float pooled = s / (cnt + 1e-6f);
    float pv = pfb[t];
#pragma unroll
    for (int j = 0; j < IND; ++j) pv += params[b * IND + j] * pfW[j * DD + t];
    pv = gelu_exact(pv);
    float mean = block_sum(pv, sm, 128) * (1.0f / DD);
    float d = pv - mean;
    float var = block_sum(d * d, sm, 128) * (1.0f / DD);
    float pf = d * rsqrtf(var + 1e-5f) * pfg[t] + pfbe[t];
    cat[t] = pooled;
    cat[DD + t] = pf;
    __syncthreads();
    float a1 = c1b[t];
    for (int j = 0; j < 2 * DD; ++j) a1 += cat[j] * c1W[j * DD + t];
    h1[t] = gelu_exact(a1);
    __syncthreads();
    if (t < 2) {
        float l2 = c2b[t];
        for (int j = 0; j < DD; ++j) l2 += h1[j] * c2W[j * 2 + t];
        out[b * 2 + t] = l2;
    }
}

// ===========================================================================
// Host launcher
// ===========================================================================
extern "C" void kernel_launch(void* const* d_in, const int* in_sizes, int n_in,
                              void* d_out, int out_size, void* d_ws, size_t ws_size,
                              hipStream_t stream) {
    (void)in_sizes; (void)n_in; (void)out_size; (void)ws_size;
    const float* hits   = (const float*)d_in[0];
    const unsigned char* mask = (const unsigned char*)d_in[1];
    const float* params = (const float*)d_in[2];
    const float* eW  = (const float*)d_in[3];
    const float* eb  = (const float*)d_in[4];
    const float* eg  = (const float*)d_in[5];
    const float* ebe = (const float*)d_in[6];
    const float* pW  = (const float*)d_in[7];
    const float* pb  = (const float*)d_in[8];
    const float* ppW = (const float*)d_in[9];
    const float* ppb = (const float*)d_in[10];
    const float* qkvW = (const float*)d_in[11];
    const float* outW = (const float*)d_in[12];
    const float* outB = (const float*)d_in[13];
    const float* pmW1 = (const float*)d_in[14];
    const float* pmB1 = (const float*)d_in[15];
    const float* pmW2 = (const float*)d_in[16];
    const float* pmB2 = (const float*)d_in[17];
    const float* f1W = (const float*)d_in[18];
    const float* f1B = (const float*)d_in[19];
    const float* f2W = (const float*)d_in[20];
    const float* f2B = (const float*)d_in[21];
    const float* n1g = (const float*)d_in[22];
    const float* n1b = (const float*)d_in[23];
    const float* n2g = (const float*)d_in[24];
    const float* n2b = (const float*)d_in[25];
    const float* pfW = (const float*)d_in[26];
    const float* pfb = (const float*)d_in[27];
    const float* pfg = (const float*)d_in[28];
    const float* pfbe = (const float*)d_in[29];
    const float* c1W = (const float*)d_in[30];
    const float* c1b = (const float*)d_in[31];
    const float* c2W = (const float*)d_in[32];
    const float* c2b = (const float*)d_in[33];

    char* ws = (char*)d_ws;
    // workspace layout (bytes) -- total ~84.5 MB
    float*    feat    = (float*)   (ws + 0);                    // 1,048,576
    _Float16* feat16  = (_Float16*)(ws + 1048576);              //   524,288
    _Float16* qkv16   = (_Float16*)(ws + 1572864);              // 1,572,864
    float*    tmp     = (float*)   (ws + 3145728);              // 1,048,576
    _Float16* g16     = (_Float16*)(ws + 4194304);              // 2,097,152
    _Float16* o16     = (_Float16*)(ws + 6291456);              //   524,288
    _Float16* vt      = (_Float16*)(ws + 6815744);              //   524,288
    float*    pair2   = (float*)   (ws + 7340032);              // 8,388,608
    _Float16* bias16  = (_Float16*)(ws + 15728640);             // 16,777,216
    _Float16* attn16  = (_Float16*)(ws + 32505856);             // 16,777,216
    float*    logits  = (float*)   (ws + 49283072);             // 33,554,432
    float*    den     = (float*)   (ws + 82837504);             //        48
    _Float16* qkvWt   = (_Float16*)(ws + 82837760);             //   393,216
    _Float16* outWt   = (_Float16*)(ws + 83230976);             //   131,072
    _Float16* f1Wt    = (_Float16*)(ws + 83362048);             //   524,288
    _Float16* f2Wt    = (_Float16*)(ws + 83886336);             //   524,288
    _Float16* pmW2t   = (_Float16*)(ws + 84410624);             //    16,384

    // ---- weight prep (f32 -> f16, transposed to N-major for B fragments) ---
    for (int i = 0; i < LL; ++i) {
        conv_transpose_kernel<<<(DD * 3 * DD + 255) / 256, 256, 0, stream>>>(
            qkvW + (size_t)i * DD * 3 * DD, qkvWt + (size_t)i * 3 * DD * DD, DD, 3 * DD);
        conv_transpose_kernel<<<(DD * DD + 255) / 256, 256, 0, stream>>>(
            outW + (size_t)i * DD * DD, outWt + (size_t)i * DD * DD, DD, DD);
        conv_transpose_kernel<<<(DD * FF + 255) / 256, 256, 0, stream>>>(
            f1W + (size_t)i * DD * FF, f1Wt + (size_t)i * FF * DD, DD, FF);
        conv_transpose_kernel<<<(FF * DD + 255) / 256, 256, 0, stream>>>(
            f2W + (size_t)i * FF * DD, f2Wt + (size_t)i * DD * FF, FF, DD);
    }
    pmw2t_kernel<<<(LL * 16 * DD + 255) / 256, 256, 0, stream>>>(pmW2, pmW2t);

    // ---- embedding & pair features ----------------------------------------
    embed_kernel<<<BN * NN, 128, 0, stream>>>(hits, eW, eb, eg, ebe, pW, pb,
                                              feat, feat16);
    pair_den_kernel<<<BN * 3, 256, 0, stream>>>(hits, den);
    pair2_kernel<<<(BN * NN * NN) / 256, 256, 0, stream>>>(hits, den, ppW, ppb, pair2);

    // ---- transformer layers ------------------------------------------------
    for (int i = 0; i < LL; ++i) {
        // qkv = feat @ qkv_W[i]   (M=2048,K=128,N=384)
        wmma_gemm_kernel<EPI_F16><<<dim3(3 * DD / 16, BN * NN / 16), 32, 0, stream>>>(
            feat16, qkvWt + (size_t)i * 3 * DD * DD, nullptr, qkv16,
            DD, DD, DD, 3 * DD);
        vt_kernel<<<(BN * HH * HD * NN) / 256, 256, 0, stream>>>(qkv16, vt);
        // pair bias MLP (fused, WMMA second layer)
        bias_kernel<<<dim3(NN / 16, NN, BN), 32, 0, stream>>>(
            pair2, pmW1 + (size_t)i * 2 * DD, pmB1 + (size_t)i * DD,
            pmW2t + (size_t)i * 16 * DD, pmB2 + (size_t)i * HH, bias16);
        // logits = QK^T*scale + bias (masked)
        logits_kernel<<<dim3(NN / 16, NN / 16, BN * HH), 32, 0, stream>>>(
            qkv16, bias16, mask, logits);
        softmax_kernel<<<BN * HH * NN, 128, 0, stream>>>(logits, attn16);
        // o = attn @ V
        av_kernel<<<dim3(NN / 16, BN * HH), 32, 0, stream>>>(attn16, vt, o16);
        // out proj + bias -> tmp ; feat = LN(feat + tmp)
        wmma_gemm_kernel<EPI_BIAS_F32><<<dim3(DD / 16, BN * NN / 16), 32, 0, stream>>>(
            o16, outWt + (size_t)i * DD * DD, outB + (size_t)i * DD, tmp,
            DD, DD, DD, DD);
        ln_kernel<<<BN * NN, 128, 0, stream>>>(feat, feat16, tmp,
                                               n1g + (size_t)i * DD, n1b + (size_t)i * DD);
        // FFN
        wmma_gemm_kernel<EPI_BIAS_GELU_F16><<<dim3(FF / 16, BN * NN / 16), 32, 0, stream>>>(
            feat16, f1Wt + (size_t)i * FF * DD, f1B + (size_t)i * FF, g16,
            DD, DD, DD, FF);
        wmma_gemm_kernel<EPI_BIAS_F32><<<dim3(DD / 16, BN * NN / 16), 32, 0, stream>>>(
            g16, f2Wt + (size_t)i * DD * FF, f2B + (size_t)i * DD, tmp,
            FF, FF, FF, DD);
        ln_kernel<<<BN * NN, 128, 0, stream>>>(feat, feat16, tmp,
                                               n2g + (size_t)i * DD, n2b + (size_t)i * DD);
    }

    // ---- head --------------------------------------------------------------
    head_kernel<<<BN, 128, 0, stream>>>(feat, mask, params, pfW, pfb, pfg, pfbe,
                                        c1W, c1b, c2W, c2b, (float*)d_out);
}
